// _NonLocalBlockND_67800353734736
// MI455X (gfx1250) — matrix operand order repaired and checked by reading, not compiled, not run on hardware
//
#include <hip/hip_runtime.h>

typedef __attribute__((ext_vector_type(16))) _Float16 v16h;
typedef __attribute__((ext_vector_type(8)))  _Float16 v8h;
typedef __attribute__((ext_vector_type(4)))  _Float16 v4h;
typedef __attribute__((ext_vector_type(8)))  float    v8f;
typedef __attribute__((ext_vector_type(4)))  float    v4f;

#define B_   8
#define C_   64
#define IC_  32
#define H_   64
#define W_   64
#define ROW  (H_*W_)   // 4096
#define DH   (H_/2)    // 32
#define DW   (W_/2)    // 32
#define COL  (DH*DW)   // 1024
#define SPAD 4         // f32 S row padding (floats)
#define PPAD 8         // f16 P row padding (halves)
#define INV_2SIG2 (1.0f/18.0f)   // sigma = max(1, 32/10) = 3 -> 2*sigma^2 = 18

// ---------------------------------------------------------------------------
// Kernel 1: theta = conv1x1(x) + b, stored PRE-SWIZZLED in WMMA A-fragment
// layout: thetaSw[(b*256 + tile)*32 + lane][16 halves]  (one v16h per lane)
//   lane < 16 : row m=lane,    K = {0..15}
//   lane >=16 : row m=lane-16, K = {16..31}
// ---------------------------------------------------------------------------
__global__ void proj_theta_kernel(const float* __restrict__ x,
                                  const float* __restrict__ tw,
                                  const float* __restrict__ tb,
                                  _Float16* __restrict__ thetaSw) {
    __shared__ float w[IC_*C_];
    __shared__ float bias[IC_];
    for (int i = threadIdx.x; i < IC_*C_; i += blockDim.x) w[i] = tw[i];
    if (threadIdx.x < IC_) bias[threadIdx.x] = tb[threadIdx.x];
    __syncthreads();

    int b = blockIdx.x / (ROW/256);
    int p = (blockIdx.x % (ROW/256))*256 + threadIdx.x;
    const float* xb = x + (size_t)b*C_*ROW;

    float acc[IC_];
    #pragma unroll
    for (int ic = 0; ic < IC_; ++ic) acc[ic] = bias[ic];
    for (int c = 0; c < C_; ++c) {
        float xv = xb[c*ROW + p];
        #pragma unroll
        for (int ic = 0; ic < IC_; ++ic) acc[ic] += w[ic*C_+c]*xv;
    }
    int tile = p >> 4, m = p & 15;
    v16h lo, hi;
    #pragma unroll
    for (int h = 0; h < 16; ++h) {
        int k = h;
        lo[h] = (_Float16)acc[k];
        hi[h] = (_Float16)acc[16 + k];
    }
    _Float16* dst = thetaSw + ((size_t)(b*(ROW/16) + tile)*32 + m)*16;
    *(v16h*)dst         = lo;   // lane m slot
    *(v16h*)(dst + 256) = hi;   // lane m+16 slot
}

// ---------------------------------------------------------------------------
// Kernel 2: g/phi conv1x1 + 2x2 maxpool.
//   phi -> fp16 (col, IC): B-fragment lanes read 16 contiguous halves.
//   g   -> fp32 (IC, col) raw, for the gnorm reduction.
// ---------------------------------------------------------------------------
__global__ void proj_pool_kernel(const float* __restrict__ x,
                                 const float* __restrict__ gw, const float* __restrict__ gb,
                                 const float* __restrict__ pw, const float* __restrict__ pb,
                                 _Float16* __restrict__ phiT, float* __restrict__ gRaw) {
    __shared__ float wg[IC_*C_], wp[IC_*C_];
    __shared__ float bg[IC_], bp[IC_];
    for (int i = threadIdx.x; i < IC_*C_; i += blockDim.x) { wg[i] = gw[i]; wp[i] = pw[i]; }
    if (threadIdx.x < IC_) { bg[threadIdx.x] = gb[threadIdx.x]; bp[threadIdx.x] = pb[threadIdx.x]; }
    __syncthreads();

    int b = blockIdx.x / (COL/256);
    int q = (blockIdx.x % (COL/256))*256 + threadIdx.x;
    int qy = q >> 5, qx = q & 31;
    const float* xb = x + (size_t)b*C_*ROW;

    float gm[IC_], pm[IC_];
    #pragma unroll
    for (int ic = 0; ic < IC_; ++ic) { gm[ic] = -3.0e38f; pm[ic] = -3.0e38f; }

    for (int d = 0; d < 4; ++d) {
        int py = 2*qy + (d >> 1), px = 2*qx + (d & 1);
        int p = py*W_ + px;
        float ga[IC_], pa[IC_];
        #pragma unroll
        for (int ic = 0; ic < IC_; ++ic) { ga[ic] = bg[ic]; pa[ic] = bp[ic]; }
        for (int c = 0; c < C_; ++c) {
            float xv = xb[c*ROW + p];
            #pragma unroll
            for (int ic = 0; ic < IC_; ++ic) {
                ga[ic] += wg[ic*C_+c]*xv;
                pa[ic] += wp[ic*C_+c]*xv;
            }
        }
        #pragma unroll
        for (int ic = 0; ic < IC_; ++ic) {
            gm[ic] = fmaxf(gm[ic], ga[ic]);
            pm[ic] = fmaxf(pm[ic], pa[ic]);
        }
    }
    v16h p0, p1;
    #pragma unroll
    for (int h = 0; h < 16; ++h) { p0[h] = (_Float16)pm[h]; p1[h] = (_Float16)pm[16+h]; }
    _Float16* pd = phiT + ((size_t)b*COL + q)*IC_;
    *(v16h*)pd        = p0;
    *(v16h*)(pd + 16) = p1;
    #pragma unroll
    for (int ic = 0; ic < IC_; ++ic)
        gRaw[(size_t)b*IC_*COL + ic*COL + q] = gm[ic];
}

// ---------------------------------------------------------------------------
// Kernel 3: per-(b,ic) minmax over col; write normalized g fp16 (IC, col)
// so GEMM-2 B-fragment lanes read 16 contiguous halves.
// ---------------------------------------------------------------------------
__global__ void gnorm_kernel(const float* __restrict__ gRaw, _Float16* __restrict__ gT) {
    int b  = blockIdx.x / IC_;
    int ic = blockIdx.x % IC_;
    const float* src = gRaw + ((size_t)b*IC_ + ic)*COL;
    __shared__ float smn[256], smx[256];
    float mn = 3.0e38f, mx = -3.0e38f;
    for (int q = threadIdx.x; q < COL; q += 256) {
        float v = src[q]; mn = fminf(mn, v); mx = fmaxf(mx, v);
    }
    smn[threadIdx.x] = mn; smx[threadIdx.x] = mx;
    __syncthreads();
    for (int s = 128; s > 0; s >>= 1) {
        if (threadIdx.x < s) {
            smn[threadIdx.x] = fminf(smn[threadIdx.x], smn[threadIdx.x+s]);
            smx[threadIdx.x] = fmaxf(smx[threadIdx.x], smx[threadIdx.x+s]);
        }
        __syncthreads();
    }
    float m0 = smn[0];
    float inv = 1.0f/(smx[0]-m0);
    _Float16* dst = gT + ((size_t)b*IC_ + ic)*COL;
    for (int q = threadIdx.x; q < COL; q += 256) dst[q] = (_Float16)((src[q]-m0)*inv);
}

// ---------------------------------------------------------------------------
// Kernel 4: fused attention for one (batch, 16-row tile). 4 waves / 128 thr.
// ---------------------------------------------------------------------------
__global__ __launch_bounds__(128)
void attn_kernel(const _Float16* __restrict__ thetaSw,
                 const _Float16* __restrict__ phiT,
                 const _Float16* __restrict__ gT,
                 float* __restrict__ yOut) {
    int b    = blockIdx.x / (ROW/16);
    int tile = blockIdx.x % (ROW/16);
    int tid  = threadIdx.x;
    int lane = tid & 31;
    int wave = tid >> 5;

    __shared__ float    S[16][COL + SPAD];   // scores (fp32)
    __shared__ _Float16 P[16][COL + PPAD];   // softmax numerators (fp16)
    __shared__ float yPart[4][16][16];
    __shared__ float redMin[16][8], redMax[16][8];
    __shared__ float rowA[16], rowB[16], rowC[16], zMax[16], invDen[16];
    __shared__ float cxA[16], cyA[16];

    int m    = lane & 15;
    int kbA  = (lane < 16) ? 0 : 8;
    int kb16 = (lane < 16) ? 0 : 16;
    int mB   = (lane < 16) ? 0 : 8;

    // ---- A fragment (theta tile): single contiguous 32B vector load ----
    v16h aTh = *(const v16h*)(thetaSw + ((size_t)(b*(ROW/16) + tile)*32 + lane)*16);

    // ---- Phase 1: S = theta @ phi over 64 col-chunks, software-pipelined ----
    const _Float16* phBase = phiT + (size_t)b*COL*IC_ + (size_t)m*IC_ + kb16;
    v16h bPhi = *(const v16h*)(phBase + (size_t)wave*16*IC_);   // prefetch chunk 0
    #pragma unroll 4
    for (int i = 0; i < 16; ++i) {
        int n0 = (wave + 4*i)*16;
        v16h bCur = bPhi;
        if (i < 15)                                              // prefetch next chunk
            bPhi = *(const v16h*)(phBase + (size_t)(n0 + 64)*IC_);
        v8f acc = {};
        acc = __builtin_amdgcn_wmma_f32_16x16x32_f16(false, aTh, false, bCur,
                                                     (short)0, acc, false, false);
        #pragma unroll
        for (int v = 0; v < 8; ++v) S[mB+v][n0+m] = acc[v];
    }
    __syncthreads();

    int r = tid >> 3, slot = tid & 7;     // 8 threads per row, contiguous 128-col span
    int cBase = slot*128;

    // ---- Phase 2a: row min/max of S (vectorized) ----
    {
        float mn = 3.0e38f, mx = -3.0e38f;
        for (int i = 0; i < 32; ++i) {
            v4f v = *(const v4f*)&S[r][cBase + 4*i];
            #pragma unroll
            for (int j = 0; j < 4; ++j) { mn = fminf(mn, v[j]); mx = fmaxf(mx, v[j]); }
        }
        redMin[r][slot] = mn; redMax[r][slot] = mx;
    }
    __syncthreads();
    if (tid < 16) {
        float mn = redMin[tid][0], mx = redMax[tid][0];
        #pragma unroll
        for (int s = 1; s < 8; ++s) { mn = fminf(mn, redMin[tid][s]); mx = fmaxf(mx, redMax[tid][s]); }
        float invF = 1.0f/(mx - mn);
        int j = tile*16 + tid;
        float cx = (float)((j >> 2) / DW);
        float cy = (float)((j >> 2) % DW);
        float dxm = fmaxf(cx, (float)(DH-1) - cx);
        float dym = fmaxf(cy, (float)(DW-1) - cy);
        float tmin = -(dxm*dxm + dym*dym) * INV_2SIG2;   // temp max = 0
        float invT = -1.0f/tmin;
        rowA[tid] = invF;
        rowB[tid] = invT;
        rowC[tid] = -mn*invF - tmin*invT - 2.0f;
        cxA[tid] = cx; cyA[tid] = cy;
    }
    __syncthreads();

    // ---- Phase 2b: z row max ----
    {
        float a_ = rowA[r], bb = rowB[r], cb = rowC[r], cx = cxA[r], cy = cyA[r];
        float mx = -3.0e38f;
        for (int i = 0; i < 32; ++i) {
            int c = cBase + 4*i;
            v4f v = *(const v4f*)&S[r][c];
            #pragma unroll
            for (int j = 0; j < 4; ++j) {
                float dx = (float)((c+j) >> 5) - cx, dy = (float)((c+j) & 31) - cy;
                float t = -(dx*dx + dy*dy) * INV_2SIG2;
                mx = fmaxf(mx, v[j]*a_ + t*bb + cb);
            }
        }
        redMax[r][slot] = mx;
    }
    __syncthreads();
    if (tid < 16) {
        float mx = redMax[tid][0];
        #pragma unroll
        for (int s = 1; s < 8; ++s) mx = fmaxf(mx, redMax[tid][s]);
        zMax[tid] = mx;
    }
    __syncthreads();

    // ---- Phase 2c: e = exp(z - zmax) -> P (fp16), accumulate row sum ----
    {
        float a_ = rowA[r], bb = rowB[r], cb = rowC[r], cx = cxA[r], cy = cyA[r], zm = zMax[r];
        float sum = 0.f;
        for (int i = 0; i < 32; ++i) {
            int c = cBase + 4*i;
            v4f v = *(const v4f*)&S[r][c];
            v4h pv;
            #pragma unroll
            for (int j = 0; j < 4; ++j) {
                float dx = (float)((c+j) >> 5) - cx, dy = (float)((c+j) & 31) - cy;
                float t = -(dx*dx + dy*dy) * INV_2SIG2;
                float e = __expf(v[j]*a_ + t*bb + cb - zm);
                pv[j] = (_Float16)e;
                sum += e;
            }
            *(v4h*)&P[r][c] = pv;
        }
        redMax[r][slot] = sum;
    }
    __syncthreads();
    if (tid < 16) {
        float s_ = 0.f;
        #pragma unroll
        for (int s = 0; s < 8; ++s) s_ += redMax[tid][s];
        invDen[tid] = 1.0f/s_;
    }
    __syncthreads();

    // ---- Phase 3: y = P @ g_n (division by sum deferred to epilogue) ----
    {
        int nc = wave & 1;                 // output IC chunk of 16
        int kh = wave >> 1;                // K half (512)
        const _Float16* gB = gT + ((size_t)b*IC_ + nc*16 + m)*COL;
        v8f acc = {};
        #pragma unroll
        for (int ks = 0; ks < 16; ++ks) {
            int kOff = kh*512 + ks*32;
            // A fragment: two contiguous 16B LDS loads
            v8h alo = *(const v8h*)&P[m][kOff + kb16];
            v8h ahi = *(const v8h*)&P[m][kOff + kb16 + 8];
            v16h aP;
            #pragma unroll
            for (int h = 0; h < 8; ++h) { aP[h] = alo[h]; aP[8+h] = ahi[h]; }
            // B fragment: contiguous 32B global load
            v16h bG = *(const v16h*)(gB + kOff + kb16);
            acc = __builtin_amdgcn_wmma_f32_16x16x32_f16(false, aP, false, bG,
                                                         (short)0, acc, false, false);
        }
        #pragma unroll
        for (int v = 0; v < 8; ++v) yPart[wave][mB+v][m] = acc[v] * invDen[mB+v];
    }
    __syncthreads();

    // ---- combine K-halves and write y (B, IC, row) fp32 ----
    for (int o = tid; o < 16*IC_; o += 128) {
        int mm = o >> 5;
        int n  = o & 31;
        int nc = n >> 4, n16 = n & 15;
        float y = yPart[nc][mm][n16] + yPart[nc+2][mm][n16];
        yOut[((size_t)b*IC_ + n)*ROW + (size_t)tile*16 + mm] = y;
    }
}

// ---------------------------------------------------------------------------
// Kernel 5: W_y = conv1x1(y, w_w)+w_b; BN; + x
// ---------------------------------------------------------------------------
__global__ void fuse_out_kernel(const float* __restrict__ x, const float* __restrict__ y,
                                const float* __restrict__ ww, const float* __restrict__ wb,
                                const float* __restrict__ gamma, const float* __restrict__ beta,
                                const float* __restrict__ mean, const float* __restrict__ var,
                                float* __restrict__ out) {
    size_t idx = (size_t)blockIdx.x*256 + threadIdx.x;
    int p  = (int)(idx % ROW);
    int bc = (int)(idx / ROW);
    int c  = bc % C_;
    int b  = bc / C_;
    const float* ys = y + (size_t)b*IC_*ROW + p;
    float acc = wb[c];
    #pragma unroll
    for (int ic = 0; ic < IC_; ++ic) acc += ww[c*IC_+ic]*ys[ic*ROW];
    out[idx] = gamma[c]*(acc - mean[c])*rsqrtf(var[c] + 1e-5f) + beta[c] + x[idx];
}

// ---------------------------------------------------------------------------
extern "C" void kernel_launch(void* const* d_in, const int* in_sizes, int n_in,
                              void* d_out, int out_size, void* d_ws, size_t ws_size,
                              hipStream_t stream) {
    (void)in_sizes; (void)n_in; (void)out_size; (void)ws_size;
    const float* x       = (const float*)d_in[0];
    const float* g_w     = (const float*)d_in[1];
    const float* g_b     = (const float*)d_in[2];
    const float* theta_w = (const float*)d_in[3];
    const float* theta_b = (const float*)d_in[4];
    const float* phi_w   = (const float*)d_in[5];
    const float* phi_b   = (const float*)d_in[6];
    const float* w_w     = (const float*)d_in[7];
    const float* w_b     = (const float*)d_in[8];
    const float* bn_g    = (const float*)d_in[9];
    const float* bn_b    = (const float*)d_in[10];
    const float* bn_m    = (const float*)d_in[11];
    const float* bn_v    = (const float*)d_in[12];

    char* ws = (char*)d_ws;
    _Float16* thetaSw = (_Float16*)(ws);                                   // 2 MB
    _Float16* phiT    = (_Float16*)(ws + (size_t)2*1024*1024);             // 0.5 MB
    _Float16* gT      = (_Float16*)(ws + (size_t)2*1024*1024 + 512*1024);  // 0.5 MB
    float*    gRaw    = (float*)   (ws + (size_t)3*1024*1024);             // 1 MB
    float*    yWs     = (float*)   (ws + (size_t)4*1024*1024);             // 4 MB

    proj_theta_kernel<<<B_*(ROW/256), 256, 0, stream>>>(x, theta_w, theta_b, thetaSw);
    proj_pool_kernel <<<B_*(COL/256), 256, 0, stream>>>(x, g_w, g_b, phi_w, phi_b, phiT, gRaw);
    gnorm_kernel     <<<B_*IC_,       256, 0, stream>>>(gRaw, gT);
    attn_kernel      <<<B_*(ROW/16),  128, 0, stream>>>(thetaSw, phiT, gT, yWs);
    fuse_out_kernel  <<<(B_*C_*ROW)/256, 256, 0, stream>>>(x, yWs, w_w, w_b,
                                                           bn_g, bn_b, bn_m, bn_v,
                                                           (float*)d_out);
}
